// TextCNNModel_74036646249160
// MI455X (gfx1250) — compile-verified, matrix operand order checked by reading, not run on hardware
//
#include <hip/hip_runtime.h>

#define VOCAB 50000
#define DEMB 128
#define CCH 128
#define NCLS 200
#define NSAMP 4096
#define SLEN 128
#define NVID 256

typedef __attribute__((ext_vector_type(16))) __bf16 v16bf;
typedef __attribute__((ext_vector_type(8)))  float v8f;

static constexpr int XS  = 136;            // padded LDS row stride (elements), 272B = 16B-aligned
static constexpr int WS  = 136;            // same for transposed weights
static constexpr int TAP = CCH * WS;       // elements per staged tap (17408)

union BF16Frag { v16bf v; uint4 q[2]; };

// Order-preserving float <-> uint key (monotonic for ALL floats incl. negatives)
__device__ __forceinline__ unsigned fkey(float f) {
    int i = __float_as_int(f);
    return (i < 0) ? ~(unsigned)i : ((unsigned)i | 0x80000000u);
}
__device__ __forceinline__ float funkey(unsigned u) {
    int i = (u & 0x80000000u) ? (int)(u & 0x7FFFFFFFu) : (int)~u;
    return __int_as_float(i);
}

__global__ __launch_bounds__(256) void init_out_kernel(float* out, int n) {
    int i = blockIdx.x * blockDim.x + threadIdx.x;
    if (i < n) out[i] = 0.0f;   // sigmoid scores are > 0, so 0 is the identity for max
}

// Pre-transpose + convert all 12 conv taps to bf16 in the exact padded LDS layout:
// wsbf[tj][c*WS + d] = W_tap[d][c], pad columns zeroed.
__global__ __launch_bounds__(256) void prep_weights_kernel(
    const float* __restrict__ W3, const float* __restrict__ W4,
    const float* __restrict__ W5, __bf16* __restrict__ wsbf) {
    int e = blockIdx.x * 256 + threadIdx.x;
    if (e >= 12 * TAP) return;
    int tj  = e / TAP;
    int rem = e - tj * TAP;
    int c   = rem / WS;
    int d   = rem - c * WS;
    const float* W; int j;
    if (tj < 3)      { W = W3; j = tj;     }
    else if (tj < 7) { W = W4; j = tj - 3; }
    else             { W = W5; j = tj - 7; }
    float v = (d < DEMB) ? W[(j * DEMB + d) * CCH + c] : 0.0f;
    wsbf[e] = (__bf16)v;
}

__global__ __launch_bounds__(256) void textcnn_kernel(
    const int* __restrict__ input_x, const int* __restrict__ segment_ids,
    const float* __restrict__ emb,
    const float* __restrict__ W3, const float* __restrict__ b3,
    const float* __restrict__ W4, const float* __restrict__ b4,
    const float* __restrict__ W5, const float* __restrict__ b5,
    const float* __restrict__ Wout, const float* __restrict__ bout,
    const __bf16* __restrict__ wsbf,     // prepped bf16 taps (nullable)
    float* __restrict__ out)
{
    __shared__ __bf16 Xs[SLEN * XS];        // 34,816 B  (sample block, bf16)
    __shared__ __bf16 Wt[5][TAP];           // 174,080 B (transposed taps: [tap][c*WS + d])
    __shared__ unsigned featkey[3 * CCH];   // 1,536 B   (per-channel time-max, keyed)

    const int n       = blockIdx.x;         // sample
    const int tid     = threadIdx.x;
    const int lane    = tid & 31;
    const int wave    = tid >> 5;           // M-tile owned by this wave
    const int laneRow = lane & 15;
    const int khalf   = lane >> 4;

    // ---- init per-channel max keys ----
    for (int i = tid; i < 3 * CCH; i += 256) featkey[i] = fkey(-INFINITY);

    // ---- gather embedding rows -> LDS (f32 -> bf16) ----
    #pragma unroll 4
    for (int i = 0; i < (SLEN * DEMB) / 256; ++i) {
        int e = tid + i * 256;
        int s = e >> 7;
        int d = e & 127;
        int idx = input_x[n * SLEN + s];
        Xs[s * XS + d] = (__bf16)emb[idx * DEMB + d];
    }
    __syncthreads();

    const float* Wk[3]      = { W3, W4, W5 };
    const float* bk[3]      = { b3, b4, b5 };
    const int    ks[3]      = { 3, 4, 5 };
    const int    tapbase[3] = { 0, 3, 7 };
    const int    m          = wave;

    for (int ci = 0; ci < 3; ++ci) {
        const int k = ks[ci];

        __syncthreads();   // previous conv finished using Wt
        if (wsbf) {
            // ---- async copy k prepped taps (already in final layout) to LDS ----
            const __bf16* src = wsbf + (size_t)tapbase[ci] * TAP;
            const int nchunk = k * (TAP / 8);         // 16B chunks
            for (int e = tid; e < nchunk; e += 256) {
                unsigned ldsoff = (unsigned)(unsigned long long)
                                  (const void*)(&Wt[0][0] + (size_t)e * 8);
                unsigned long long ga = (unsigned long long)(const void*)(src + (size_t)e * 8);
                asm volatile("global_load_async_to_lds_b128 %0, %1, off"
                             :: "v"(ldsoff), "v"(ga) : "memory");
            }
            asm volatile("s_wait_asynccnt 0x0" ::: "memory");
        } else {
            // ---- fallback: stage from f32, transpose + convert ----
            for (int e = tid; e < k * DEMB * CCH; e += 256) {
                int j   = e >> 14;
                int rem = e & 16383;
                int d   = rem >> 7;
                int c   = rem & 127;
                Wt[j][c * WS + d] = (__bf16)Wk[ci][(j * DEMB + d) * CCH + c];
            }
        }
        __syncthreads();

        for (int nt = 0; nt < 8; ++nt) {
            v8f acc = {};
            for (int j = 0; j < k; ++j) {
                int arow = m * 16 + laneRow + j;
                if (arow > SLEN - 1) arow = SLEN - 1;   // clamped rows are masked below
                const __bf16* aptr = &Xs[arow * XS];
                const __bf16* bptr = &Wt[j][(nt * 16 + laneRow) * WS];
                #pragma unroll
                for (int kk = 0; kk < 4; ++kk) {
                    // 16-bit fragment: VGPRs 0-3 = 8 contiguous K at kk*32+khalf*8,
                    //                  VGPRs 4-7 = 8 contiguous K at +16  -> 2x ds_load_b128
                    BF16Frag a, b;
                    const __bf16* ap = aptr + kk * 32 + khalf * 8;
                    const __bf16* bp = bptr + kk * 32 + khalf * 8;
                    a.q[0] = *(const uint4*)(ap);
                    a.q[1] = *(const uint4*)(ap + 16);
                    b.q[0] = *(const uint4*)(bp);
                    b.q[1] = *(const uint4*)(bp + 16);
                    acc = __builtin_amdgcn_wmma_f32_16x16x32_bf16(
                        false, a.v, false, b.v, (short)0, acc, false, false);
                }
            }
            // ---- masked max over the 16 time rows of this tile ----
            float vmax = -INFINITY;
            #pragma unroll
            for (int r = 0; r < 8; ++r) {
                int t = m * 16 + r + khalf * 8;          // C/D layout: lanes16-31 hold M=r+8
                float x = (t <= SLEN - k) ? acc[r] : -INFINITY;
                vmax = fmaxf(vmax, x);
            }
            vmax = fmaxf(vmax, __shfl_xor(vmax, 16, 32));
            if (lane < 16) {
                int ch = nt * 16 + laneRow;
                float fv = vmax + bk[ci][ch];            // bias commutes with time-max
                atomicMax(&featkey[ci * CCH + ch], fkey(fv));
            }
        }
    }
    __syncthreads();

    // ---- classifier head + sigmoid + segment-max ----
    if (tid < NCLS) {
        float a = bout[tid];
        #pragma unroll 4
        for (int i = 0; i < 3 * CCH; ++i)
            a += funkey(featkey[i]) * Wout[i * NCLS + tid];
        float s = 1.0f / (1.0f + __expf(-a));
        int vid = segment_ids[n];
        // scores in (0,1): positive-float bit pattern is monotonic under uint max
        atomicMax((unsigned*)&out[vid * NCLS + tid], __float_as_uint(s));
    }
}

extern "C" void kernel_launch(void* const* d_in, const int* in_sizes, int n_in,
                              void* d_out, int out_size, void* d_ws, size_t ws_size,
                              hipStream_t stream) {
    const int*   input_x     = (const int*)  d_in[0];
    const int*   segment_ids = (const int*)  d_in[1];
    const float* emb         = (const float*)d_in[2];
    const float* W3          = (const float*)d_in[3];
    const float* b3          = (const float*)d_in[4];
    const float* W4          = (const float*)d_in[5];
    const float* b4          = (const float*)d_in[6];
    const float* W5          = (const float*)d_in[7];
    const float* b5          = (const float*)d_in[8];
    const float* Wout        = (const float*)d_in[9];
    const float* bout        = (const float*)d_in[10];
    float* out = (float*)d_out;

    const size_t ws_needed = (size_t)12 * TAP * sizeof(__bf16);
    __bf16* wsbf = (ws_size >= ws_needed) ? (__bf16*)d_ws : nullptr;

    if (wsbf)
        prep_weights_kernel<<<(12 * TAP + 255) / 256, 256, 0, stream>>>(W3, W4, W5, wsbf);
    init_out_kernel<<<(NVID * NCLS + 255) / 256, 256, 0, stream>>>(out, NVID * NCLS);
    textcnn_kernel<<<NSAMP, 256, 0, stream>>>(
        input_x, segment_ids, emb, W3, b3, W4, b4, W5, b5, Wout, bout, wsbf, out);
}
